// ContrastiveLoss_78228534329345
// MI455X (gfx1250) — matrix operand order repaired and checked by reading, not compiled, not run on hardware
//
#include <hip/hip_runtime.h>
#include <hip/hip_bf16.h>

// ---------------------------------------------------------------------------
// Contrastive loss for MI455X (gfx1250, wave32, WMMA, async-to-LDS DMA).
//
//   1) cl_norm_split : L2-normalize rows, split fp32 -> bf16 hi + bf16 lo.
//   2) cl_logits_tile: 128x128 logit tiles, v_wmma_f32_16x16x32_bf16 with
//                      3 WMMAs per K-step (hi*hi + hi*lo + lo*hi).
//                      Double-buffered LDS filled by
//                      GLOBAL_LOAD_ASYNC_TO_LDS_B128 (ASYNCcnt), overlapping
//                      global traffic with the WMMA stream. Fused epilogue:
//                      exp, deterministic row/col partial sums, diagonal.
//   3) cl_finalize   : reduce partials to the two scalar losses.
// ---------------------------------------------------------------------------

typedef __attribute__((ext_vector_type(16))) __bf16 v16bf;
typedef __attribute__((ext_vector_type(8)))  __bf16 v8bf;
typedef __attribute__((ext_vector_type(8)))  float  v8f;

static constexpr int   N_ROWS = 4096;
static constexpr int   DIM    = 1024;
static constexpr float INV_T  = 10.0f;   // 1 / TEMPERATURE

static constexpr int BM   = 128;
static constexpr int BN   = 128;
static constexpr int BK   = 32;
static constexpr int LDSS = 40;                 // padded LDS row stride (bf16)
static constexpr int NTILES = N_ROWS / BM;      // 32
static constexpr int KSTEPS = DIM / BK;         // 32

// Dynamic-LDS byte layout (no static __shared__ in the GEMM kernel, so the
// dynamic region starts at LDS byte offset 0 and async-DMA LDS addresses are
// plain byte offsets).
static constexpr unsigned TILE_E    = BM * LDSS;        // elems per tile buf
static constexpr unsigned TILE_B    = TILE_E * 2;       // 10240 B
static constexpr unsigned BUF_B     = 4 * TILE_B;       // Ah,Al,Bh,Bl = 40960 B
static constexpr unsigned COLBUF_B  = 8 * BN * 4;       // 4096 B
static constexpr unsigned LDS_TOTAL = 2 * BUF_B + COLBUF_B; // 86016 B

// ------------------- async copy / wait helpers -----------------------------
__device__ __forceinline__ void async_b128(unsigned lds_byte,
                                           const void* gbase,
                                           unsigned gbyte_off) {
  // GVS form: mem = SGPR64 base + VGPR 32-bit byte offset; LDS dst = VGPR.
  asm volatile("global_load_async_to_lds_b128 %0, %1, %2"
               :
               : "v"(lds_byte), "v"(gbyte_off), "s"(gbase)
               : "memory");
}

__device__ __forceinline__ void wait_async_le8() {
#if defined(__has_builtin) && __has_builtin(__builtin_amdgcn_s_wait_asynccnt)
  __builtin_amdgcn_s_wait_asynccnt(8);
#else
  asm volatile("s_wait_asynccnt 8" ::: "memory");
#endif
}

__device__ __forceinline__ void wait_async_0() {
#if defined(__has_builtin) && __has_builtin(__builtin_amdgcn_s_wait_asynccnt)
  __builtin_amdgcn_s_wait_asynccnt(0);
#else
  asm volatile("s_wait_asynccnt 0" ::: "memory");
#endif
}

// Issue this thread's 8 x b128 async copies for one 4-tile K-slab.
__device__ __forceinline__ void issue_slab(const __bf16* Ahi, const __bf16* Alo,
                                           const __bf16* Bhi, const __bf16* Blo,
                                           unsigned gA, unsigned gB,
                                           unsigned ldsBase, unsigned ltb) {
  async_b128(ldsBase + 0u * TILE_B + ltb,       Ahi, gA);
  async_b128(ldsBase + 0u * TILE_B + ltb + 16u, Ahi, gA + 16u);
  async_b128(ldsBase + 1u * TILE_B + ltb,       Alo, gA);
  async_b128(ldsBase + 1u * TILE_B + ltb + 16u, Alo, gA + 16u);
  async_b128(ldsBase + 2u * TILE_B + ltb,       Bhi, gB);
  async_b128(ldsBase + 2u * TILE_B + ltb + 16u, Bhi, gB + 16u);
  async_b128(ldsBase + 3u * TILE_B + ltb,       Blo, gB);
  async_b128(ldsBase + 3u * TILE_B + ltb + 16u, Blo, gB + 16u);
}

// ------------------- kernel 1: normalize + bf16 hi/lo split ----------------
__global__ __launch_bounds__(256)
void cl_norm_split(const float* __restrict__ e1, const float* __restrict__ e2,
                   __bf16* __restrict__ hi1, __bf16* __restrict__ lo1,
                   __bf16* __restrict__ hi2, __bf16* __restrict__ lo2) {
  int row = blockIdx.x;
  const float* src;
  __bf16 *hi, *lo;
  if (row < N_ROWS) {
    src = e1 + (size_t)row * DIM;
    hi  = hi1 + (size_t)row * DIM;
    lo  = lo1 + (size_t)row * DIM;
  } else {
    row -= N_ROWS;
    src = e2 + (size_t)row * DIM;
    hi  = hi2 + (size_t)row * DIM;
    lo  = lo2 + (size_t)row * DIM;
  }

  __shared__ float red[256];
  float ss = 0.f;
  for (int k = threadIdx.x; k < DIM; k += 256) {
    float v = src[k];
    ss += v * v;
  }
  red[threadIdx.x] = ss;
  __syncthreads();
  for (int s = 128; s > 0; s >>= 1) {
    if ((int)threadIdx.x < s) red[threadIdx.x] += red[threadIdx.x + s];
    __syncthreads();
  }
  const float inv = 1.0f / fmaxf(sqrtf(red[0]), 1e-12f);

  for (int k = threadIdx.x; k < DIM; k += 256) {
    float y  = src[k] * inv;
    __bf16 h = (__bf16)y;
    hi[k] = h;
    lo[k] = (__bf16)(y - (float)h);
  }
}

// ------------------- kernel 2: tiled WMMA GEMM + fused epilogue ------------
__global__ __launch_bounds__(256)
void cl_logits_tile(const __bf16* __restrict__ Ahi, const __bf16* __restrict__ Alo,
                    const __bf16* __restrict__ Bhi, const __bf16* __restrict__ Blo,
                    float* __restrict__ row_partial,   // [NTILES][N_ROWS]
                    float* __restrict__ col_partial,   // [NTILES][N_ROWS]
                    float* __restrict__ diag_logit) {  // [N_ROWS]
  extern __shared__ char smem[];   // LDS_TOTAL bytes; starts at LDS offset 0

  const int bx      = blockIdx.x;       // tile column index
  const int by      = blockIdx.y;       // tile row index
  const int rowBase = by * BM;
  const int colBase = bx * BN;

  const int tid   = threadIdx.x;
  const int wave  = tid >> 5;           // wave w owns rows [16w, 16w+16)
  const int lane  = tid & 31;
  const int lhalf = lane >> 4;          // 0 or 1
  const int lmod  = lane & 15;

  v8f acc[8];
#pragma unroll
  for (int j = 0; j < 8; ++j)
#pragma unroll
    for (int r = 0; r < 8; ++r) acc[j][r] = 0.f;

  // loader mapping: 256 threads, thread -> (row, 16-elem chunk)
  const int lr = tid >> 1;              // 0..127 tile row
  const int lc = (tid & 1) * 16;        // 0 or 16 (bf16 elems)

  const unsigned ltb    = (unsigned)(lr * LDSS + lc) * 2u;          // LDS bytes
  const unsigned gAbase = (unsigned)(((rowBase + lr) * DIM + lc) * 2);
  const unsigned gBbase = (unsigned)(((colBase + lr) * DIM + lc) * 2);

  // prologue: fill buffer 0 with K-slab 0
  issue_slab(Ahi, Alo, Bhi, Blo, gAbase, gBbase, 0u, ltb);

  for (int ks = 0; ks < KSTEPS; ++ks) {
    const int cur = ks & 1;

    __syncthreads();   // WAR: everyone done reading buf cur^1 (prev compute)
    if (ks + 1 < KSTEPS) {
      const unsigned koff = (unsigned)((ks + 1) * BK * 2);
      issue_slab(Ahi, Alo, Bhi, Blo, gAbase + koff, gBbase + koff,
                 (unsigned)(cur ^ 1) * BUF_B, ltb);
      wait_async_le8();   // first 8 outstanding (current buffer) are done
    } else {
      wait_async_0();     // drain everything
    }
    __syncthreads();      // all waves' current-buffer data is in LDS

    const __bf16* sAh = (const __bf16*)(smem + (size_t)cur * BUF_B);
    const __bf16* sAl = sAh + TILE_E;
    const __bf16* sBh = sAh + 2 * TILE_E;
    const __bf16* sBl = sAh + 3 * TILE_E;

    // A fragment (16x32 bf16), lane layout (matches B):
    //   lanes 0-15 : elems 0..15 = K 0..15
    //   lanes 16-31: elems 0..15 = K 16..31
    const int arow = wave * 16 + lmod;
    v8bf a0h = *(const v8bf*)&sAh[arow * LDSS + lhalf * 16];
    v8bf a1h = *(const v8bf*)&sAh[arow * LDSS + lhalf * 16 + 8];
    v8bf a0l = *(const v8bf*)&sAl[arow * LDSS + lhalf * 16];
    v8bf a1l = *(const v8bf*)&sAl[arow * LDSS + lhalf * 16 + 8];
    v16bf ah = __builtin_shufflevector(a0h, a1h, 0,1,2,3,4,5,6,7,8,9,10,11,12,13,14,15);
    v16bf al = __builtin_shufflevector(a0l, a1l, 0,1,2,3,4,5,6,7,8,9,10,11,12,13,14,15);

#pragma unroll
    for (int j = 0; j < 8; ++j) {
      // B fragment (32x16 bf16): lane = column N; lanes 0-15 K 0..15,
      // lanes 16-31 K 16..31 (one contiguous 16-elem chunk).
      const int brow = j * 16 + lmod;
      v16bf bh = *(const v16bf*)&sBh[brow * LDSS + lhalf * 16];
      v16bf bl = *(const v16bf*)&sBl[brow * LDSS + lhalf * 16];

      acc[j] = __builtin_amdgcn_wmma_f32_16x16x32_bf16(
          false, ah, false, bh, (short)0, acc[j], false, false);
      acc[j] = __builtin_amdgcn_wmma_f32_16x16x32_bf16(
          false, ah, false, bl, (short)0, acc[j], false, false);
      acc[j] = __builtin_amdgcn_wmma_f32_16x16x32_bf16(
          false, al, false, bh, (short)0, acc[j], false, false);
    }
  }

  // ----------------- fused epilogue: exp + partial sums --------------------
  // C/D layout: VGPR r -> M = r + 8*lhalf, column N = lmod (+ 16*j).
  float* colbuf = (float*)(smem + 2 * BUF_B);   // [8][BN]
  const bool isDiag = (bx == by);
  float rowpart[8] = {0.f, 0.f, 0.f, 0.f, 0.f, 0.f, 0.f, 0.f};

#pragma unroll
  for (int j = 0; j < 8; ++j) {
    float cs = 0.f;
#pragma unroll
    for (int r = 0; r < 8; ++r) {
      float lg = acc[j][r] * INV_T;
      float ex = __expf(lg);
      cs += ex;                 // this lane's 8 rows of one column
      rowpart[r] += ex;         // accumulate across the 8 column subtiles
    }
    cs += __shfl_xor(cs, 16, 32);   // combine the two 8-row halves
    if (lane < 16) colbuf[wave * BN + j * 16 + lane] = cs;

    // diagonal logit lives in subtile j == wave at local m == n
    if (isDiag && j == wave && lhalf == (lmod >> 3)) {
      diag_logit[rowBase + wave * 16 + lmod] = acc[j][lmod & 7] * INV_T;
    }
  }

  // reduce each row across its 16 column lanes (masks 1,2,4,8 stay in-half)
#pragma unroll
  for (int r = 0; r < 8; ++r) {
    float v = rowpart[r];
    v += __shfl_xor(v, 1, 32);
    v += __shfl_xor(v, 2, 32);
    v += __shfl_xor(v, 4, 32);
    v += __shfl_xor(v, 8, 32);
    if (lmod == 0) {
      row_partial[(size_t)bx * N_ROWS + rowBase + wave * 16 + lhalf * 8 + r] = v;
    }
  }

  __syncthreads();
  // combine the 8 waves' column sums in a fixed order (deterministic)
  if (tid < BN) {
    float s = 0.f;
#pragma unroll
    for (int w = 0; w < 8; ++w) s += colbuf[w * BN + tid];
    col_partial[(size_t)by * N_ROWS + colBase + tid] = s;
  }
}

// ------------------- kernel 3: finalize ------------------------------------
__global__ __launch_bounds__(256)
void cl_finalize(const float* __restrict__ row_partial,
                 const float* __restrict__ col_partial,
                 const float* __restrict__ diag_logit,
                 float* __restrict__ out) {
  __shared__ float s1[256];
  __shared__ float s2[256];
  float a = 0.f, b = 0.f;
  for (int i = threadIdx.x; i < N_ROWS; i += 256) {
    float rs = 0.f, cs = 0.f;
    for (int t = 0; t < NTILES; ++t) {
      rs += row_partial[(size_t)t * N_ROWS + i];
      cs += col_partial[(size_t)t * N_ROWS + i];
    }
    const float d  = diag_logit[i];
    const float ed = __expf(d);
    a += d - __logf(rs - ed);   // denominators exclude the diagonal
    b += d - __logf(cs - ed);
  }
  s1[threadIdx.x] = a;
  s2[threadIdx.x] = b;
  __syncthreads();
  for (int s = 128; s > 0; s >>= 1) {
    if ((int)threadIdx.x < s) {
      s1[threadIdx.x] += s1[threadIdx.x + s];
      s2[threadIdx.x] += s2[threadIdx.x + s];
    }
    __syncthreads();
  }
  if (threadIdx.x == 0) {
    out[0] = -s1[0];
    out[1] = -s2[0];
  }
}

// ------------------- host-side launch --------------------------------------
extern "C" void kernel_launch(void* const* d_in, const int* in_sizes, int n_in,
                              void* d_out, int out_size, void* d_ws, size_t ws_size,
                              hipStream_t stream) {
  (void)in_sizes; (void)n_in; (void)out_size; (void)ws_size;

  const float* e1  = (const float*)d_in[0];
  const float* e2  = (const float*)d_in[1];
  float*       out = (float*)d_out;

  const size_t nd = (size_t)N_ROWS * DIM;
  __bf16* Ahi = (__bf16*)d_ws;
  __bf16* Alo = Ahi + nd;
  __bf16* Bhi = Alo + nd;
  __bf16* Blo = Bhi + nd;
  float* row_partial = (float*)(Blo + nd);                       // 32*4096 f32
  float* col_partial = row_partial + (size_t)NTILES * N_ROWS;    // 32*4096 f32
  float* diag_logit  = col_partial + (size_t)NTILES * N_ROWS;    // 4096 f32

  cl_norm_split<<<2 * N_ROWS, 256, 0, stream>>>(e1, e2, Ahi, Alo, Bhi, Blo);
  cl_logits_tile<<<dim3(NTILES, NTILES), 256, LDS_TOTAL, stream>>>(
      Ahi, Alo, Bhi, Blo, row_partial, col_partial, diag_logit);
  cl_finalize<<<1, 256, 0, stream>>>(row_partial, col_partial, diag_logit, out);
}
